// MultipleNegativesRankingLoss_81123342286901
// MI455X (gfx1250) — matrix operand order repaired and checked by reading, not compiled, not run on hardware
//
#include <hip/hip_runtime.h>
#include <hip/hip_bf16.h>
#include <math.h>

// ---------------------------------------------------------------------------
// MultipleNegativesRankingLoss on MI455X (gfx1250, wave32, WMMA)
//   loss = mean_b [ logsumexp_n(20 * cos(a_b, p_n)) - 20 * cos(a_b, p_b) ]
//
// K1: row L2-normalize f32 -> f16 into workspace (A16, P16)
// K2: diag[b] = 20 * dot(A16[b], P16[b])   (keeps all conditionals out of K3)
// K3: f16 WMMA GEMM: A tile (16x1024) fully register-resident per wave,
//     2 N-tiles per pass, B software-pipelined TWO K-steps ahead (3 rotating
//     buffers); fixed-max logsumexp (scores bounded in [-20,20] ->
//     logZ = 20 + log(sum exp(v-20)) is exactly stable).
// ---------------------------------------------------------------------------

typedef __attribute__((ext_vector_type(16))) _Float16 v16h;
typedef __attribute__((ext_vector_type(8)))  _Float16 v8h;
typedef __attribute__((ext_vector_type(4)))  _Float16 v4h;
typedef __attribute__((ext_vector_type(8)))  float    v8f;

#define DIM      1024
#define SCALE_F  20.0f
#define MAXLOGIT 20.0f   // hard upper bound of 20*cosine

union Frag { v16h v; v8h h[2]; };

static __device__ __forceinline__ v8f wmma_f16(const Frag& a, const Frag& b, v8f c) {
  return __builtin_amdgcn_wmma_f32_16x16x32_f16(
      /*neg_a=*/false, a.v, /*neg_b=*/false, b.v,
      /*c_mod=*/(short)0, c, /*reuse_a=*/false, /*reuse_b=*/false);
}

// ---------------- Kernel 1: per-row L2 normalize, f32 -> f16 ----------------
__global__ __launch_bounds__(256) void mnrl_normalize_f16(
    const float* __restrict__ x, _Float16* __restrict__ y) {
  const int row = blockIdx.x;
  const int tid = threadIdx.x;

  const float4 v = ((const float4*)(x + (size_t)row * DIM))[tid];
  float ss = v.x * v.x + v.y * v.y + v.z * v.z + v.w * v.w;

  #pragma unroll
  for (int m = 1; m < 32; m <<= 1) ss += __shfl_xor(ss, m);

  __shared__ float red[8];
  if ((tid & 31) == 0) red[tid >> 5] = ss;
  __syncthreads();
  const float tot = red[0] + red[1] + red[2] + red[3] +
                    red[4] + red[5] + red[6] + red[7];
  const float rn = 1.0f / fmaxf(sqrtf(tot), 1e-8f);  // torch eps clamp

  v4h o;
  o[0] = (_Float16)(v.x * rn);
  o[1] = (_Float16)(v.y * rn);
  o[2] = (_Float16)(v.z * rn);
  o[3] = (_Float16)(v.w * rn);
  ((v4h*)(y + (size_t)row * DIM))[tid] = o;
}

// --------- Kernel 2: diag[b] = 20 * dot(A16[b,:], P16[b,:]) ----------------
__global__ __launch_bounds__(256) void mnrl_diag(
    const _Float16* __restrict__ A16, const _Float16* __restrict__ P16,
    float* __restrict__ diagG) {
  const int row = blockIdx.x;
  const int tid = threadIdx.x;

  const v4h a = ((const v4h*)(A16 + (size_t)row * DIM))[tid];
  const v4h p = ((const v4h*)(P16 + (size_t)row * DIM))[tid];
  float d = (float)a[0] * (float)p[0] + (float)a[1] * (float)p[1] +
            (float)a[2] * (float)p[2] + (float)a[3] * (float)p[3];

  #pragma unroll
  for (int m = 1; m < 32; m <<= 1) d += __shfl_xor(d, m);

  __shared__ float red[8];
  if ((tid & 31) == 0) red[tid >> 5] = d;
  __syncthreads();
  if (tid == 0) {
    const float tot = red[0] + red[1] + red[2] + red[3] +
                      red[4] + red[5] + red[6] + red[7];
    diagG[row] = SCALE_F * tot;
  }
}

// ------ Kernel 3: register-resident-A WMMA GEMM + fixed-max logsumexp ------
// Block = 256 threads = 8 waves; block owns 16 rows. Each wave keeps the full
// 16x1024 f16 A tile in 32 register fragments and stripes 128 column
// tile-pairs; per K-step: 4 global b128 loads (B, issued 2 steps early via a
// 3-deep rotating buffer) + 2 WMMAs.
__global__ __launch_bounds__(256) void mnrl_gemm_lse(
    const _Float16* __restrict__ A16, const _Float16* __restrict__ P16,
    const float* __restrict__ diagG, float* __restrict__ out, const int B) {
  __shared__ float sS[8][16];         // per-wave sum-exp per row

  const int tid  = threadIdx.x;
  const int wave = tid >> 5;
  const int lane = tid & 31;
  const int lo   = lane & 15;         // column index / row selector
  const int hi   = lane >> 4;         // K-chunk selector
  const int rowBase = blockIdx.x * 16;

  // A fragment layout: row = lo; halves 0-15 = K + 16*hi + 0..15
  // (contiguous, matches B). Load all 32 K-fragments into registers.
  const _Float16* aptr = A16 + (size_t)(rowBase + lo) * DIM + hi * 16;
  Frag aF[32];
  #pragma unroll
  for (int kk = 0; kk < 32; ++kk) {
    aF[kk].h[0] = *(const v8h*)(aptr + kk * 32);
    aF[kk].h[1] = *(const v8h*)(aptr + kk * 32 + 8);
  }

  float s[8];
  #pragma unroll
  for (int r = 0; r < 8; ++r) s[r] = 0.0f;

  const int nPairs = B >> 5;          // 128 tile-pairs

  for (int pj = wave; pj < nPairs; pj += 8) {
    // B fragment: lane = column n, contiguous K halves (hi picks K 0-15/16-31).
    const _Float16* brow0 = P16 + (size_t)(pj * 32 + lo) * DIM + hi * 16;
    const _Float16* brow1 = brow0 + (size_t)16 * DIM;

    v8f c0 = {}, c1 = {};

    // Distance-2 software pipeline: 3 rotating B buffers (constant indices
    // after full unroll -> stays entirely in registers).
    Frag f0[3], f1[3];
    #pragma unroll
    for (int p = 0; p < 2; ++p) {            // preload K-steps 0 and 1
      const int ks = p * 32;
      f0[p].h[0] = *(const v8h*)(brow0 + ks);
      f0[p].h[1] = *(const v8h*)(brow0 + ks + 8);
      f1[p].h[0] = *(const v8h*)(brow1 + ks);
      f1[p].h[1] = *(const v8h*)(brow1 + ks + 8);
    }

    #pragma unroll
    for (int kk = 0; kk < 32; ++kk) {
      if (kk + 2 < 32) {                     // prefetch K-step kk+2
        const int ks  = (kk + 2) * 32;
        const int buf = (kk + 2) % 3;
        f0[buf].h[0] = *(const v8h*)(brow0 + ks);
        f0[buf].h[1] = *(const v8h*)(brow0 + ks + 8);
        f1[buf].h[0] = *(const v8h*)(brow1 + ks);
        f1[buf].h[1] = *(const v8h*)(brow1 + ks + 8);
      }
      c0 = wmma_f16(aF[kk], f0[kk % 3], c0);
      c1 = wmma_f16(aF[kk], f1[kk % 3], c1);
    }

    // Epilogue: pure exp+accumulate (no max tracking, no conditionals).
    #pragma unroll
    for (int r = 0; r < 8; ++r) {
      s[r] += __expf(SCALE_F * c0[r] - MAXLOGIT) +
              __expf(SCALE_F * c1[r] - MAXLOGIT);
    }
  }

  // One 16-lane reduction per row (both halves reduce in parallel).
  #pragma unroll
  for (int r = 0; r < 8; ++r) {
    s[r] += __shfl_xor(s[r], 1);
    s[r] += __shfl_xor(s[r], 2);
    s[r] += __shfl_xor(s[r], 4);
    s[r] += __shfl_xor(s[r], 8);
  }
  if (lane == 0) {
    #pragma unroll
    for (int r = 0; r < 8; ++r) sS[wave][r] = s[r];          // rows 0-7
  } else if (lane == 16) {
    #pragma unroll
    for (int r = 0; r < 8; ++r) sS[wave][8 + r] = s[r];      // rows 8-15
  }
  __syncthreads();

  // Merge waves, finish logsumexp, subtract diag, reduce 16 rows, accumulate.
  if (tid < 16) {
    float S = 0.0f;
    #pragma unroll
    for (int w = 0; w < 8; ++w) S += sS[w][tid];
    float part = MAXLOGIT + __logf(S) - diagG[rowBase + tid];
    part += __shfl_xor(part, 1);
    part += __shfl_xor(part, 2);
    part += __shfl_xor(part, 4);
    part += __shfl_xor(part, 8);
    if (tid == 0) atomicAdd(out, part / (float)B);
  }
}

// ---------------------------------------------------------------------------
extern "C" void kernel_launch(void* const* d_in, const int* in_sizes, int n_in,
                              void* d_out, int out_size, void* d_ws, size_t ws_size,
                              hipStream_t stream) {
  (void)n_in; (void)out_size; (void)ws_size;
  const float* anchor  = (const float*)d_in[0];
  const float* posnegs = (const float*)d_in[1];
  float* out = (float*)d_out;

  const int B = in_sizes[0] / DIM;          // 4096

  _Float16* A16   = (_Float16*)d_ws;                  // 8 MB
  _Float16* P16   = A16 + (size_t)B * DIM;            // 8 MB
  float*    diagG = (float*)(P16 + (size_t)B * DIM);  // 16 KB

  mnrl_normalize_f16<<<B, 256, 0, stream>>>(anchor,  A16);
  mnrl_normalize_f16<<<B, 256, 0, stream>>>(posnegs, P16);
  mnrl_diag<<<B, 256, 0, stream>>>(A16, P16, diagG);
  hipMemsetAsync(d_out, 0, sizeof(float), stream);
  mnrl_gemm_lse<<<B / 16, 256, 0, stream>>>(A16, P16, diagG, out, B);
}
